// HeadMaskAttnMixture_33346126086547
// MI455X (gfx1250) — matrix operand order repaired and checked by reading, not compiled, not run on hardware
//
#include <hip/hip_runtime.h>
#include <hip/hip_bf16.h>

#define NTOK   577
#define NHEADS 12
#define HD     64
#define CD     2304   // 3*12*64
#define TT     30     // table rows
#define SGRID  24     // sqrt(576)
#define COLS   608    // 19 pairs * 32 cols (577 padded)
#define NPAIR  19

typedef __attribute__((ext_vector_type(16))) __bf16 v16bf;
typedef __attribute__((ext_vector_type(8)))  float  v8f;

static __device__ inline v8f wmma_bf16(v16bf a, v16bf b, v8f c) {
  return __builtin_amdgcn_wmma_f32_16x16x32_bf16(false, a, false, b, (short)0, c, false, false);
}

static __device__ inline int clampi(int v, int lo, int hi) {
  return v < lo ? lo : (v > hi ? hi : v);
}

// iv/ih per reference _rel_indices (row/col 0 are CLS padding -> index 0)
static __device__ inline void rel_iv_ih(int n, int m, int& iv, int& ih) {
  if (n == 0 || m == 0) { iv = 0; ih = 0; return; }
  int rn = n - 1, rm = m - 1;
  int dv = rm / SGRID - rn / SGRID;
  int dh = rm % SGRID - rn % SGRID;
  iv = clampi(dv, -14, 14) + 15;
  ih = clampi(dh, -14, 14) + 15;
}

__global__ __launch_bounds__(128)
void HeadMaskAttnMixture_kernel(const float* __restrict__ x,
                                const float* __restrict__ weights,
                                const float* __restrict__ tkv,
                                const float* __restrict__ tkh,
                                const float* __restrict__ tvv,
                                const float* __restrict__ tvh,
                                float* __restrict__ out)
{
  __shared__ float s_lds[16 * COLS];    // scores -> probs (38912 B)
  __shared__ float qtv[16 * 32];        // q . rel_k_table_v^T
  __shared__ float qth[16 * 32];
  __shared__ float av[16 * 32];         // prob histograms over iv / ih
  __shared__ float ah[16 * 32];
  __shared__ float opart[4 * 16 * HD];  // per-wave PV partials
  __shared__ float red[16 * 8];
  __shared__ float rowsum[16];

  const int tid  = threadIdx.x;
  const int lane = tid & 31;
  const int wv   = tid >> 5;          // wave 0..3
  const int lo16 = lane & 15;
  const int hi   = (lane >> 4) & 1;

  const int i0 = blockIdx.x * 16;     // q-row tile base
  const int h  = blockIdx.y;
  const int b  = blockIdx.z;

  const float w0 = weights[0], w1 = weights[1], w2 = weights[2];
  const float wh = (h < 8 ? w0 : 0.0f) + (h < 10 ? w1 : 0.0f) + w2; // head weight
  const float qscale = wh * 0.125f;   // fold w and 64^-0.5 into q

  for (int i = tid; i < 16 * 32; i += 128) { av[i] = 0.0f; ah[i] = 0.0f; }

  // ---- Phase A: Q tile -> two A fragments (16x32 bf16), scaled by w*SCALE ----
  int qrow = i0 + lo16; if (qrow > 576) qrow = 576;
  const float* qbase = x + ((size_t)b * NTOK + qrow) * CD + h * HD;
  v16bf aq0, aq1;
  {
    const float* p0 = qbase + hi * 16;        // kd 0..31 runs
    const float* p1 = qbase + 32 + hi * 16;   // kd 32..63 runs
#pragma unroll
    for (int e = 0; e < 8; ++e) {
      aq0[e]     = (__bf16)(p0[e]      * qscale);
      aq0[e + 8] = (__bf16)(p0[e + 8] * qscale);
      aq1[e]     = (__bf16)(p1[e]      * qscale);
      aq1[e + 8] = (__bf16)(p1[e + 8] * qscale);
    }
  }

  // ---- Phase B: qT = Qs * table^T for both rel_k tables (wave 0) ----
  if (wv == 0) {
#pragma unroll
    for (int tbl = 0; tbl < 2; ++tbl) {
      const float* tp  = (tbl == 0) ? tkv : tkh;
      float*       dst = (tbl == 0) ? qtv : qth;
#pragma unroll
      for (int t0 = 0; t0 < 32; t0 += 16) {
        int   t    = t0 + lo16;                  // B-matrix column (table row)
        float msk  = (t < TT) ? 1.0f : 0.0f;     // branch-free masking
        const float* trow = tp + (size_t)(t < TT ? t : 0) * HD;
        v8f c = {0.f, 0.f, 0.f, 0.f, 0.f, 0.f, 0.f, 0.f};
#pragma unroll
        for (int ks = 0; ks < 2; ++ks) {
          const float* tb = trow + ks * 32 + hi * 16;  // contiguous K-run
          v16bf bt;
#pragma unroll
          for (int e = 0; e < 16; ++e) bt[e] = (__bf16)(tb[e] * msk);
          c = wmma_bf16(ks == 0 ? aq0 : aq1, bt, c);
        }
#pragma unroll
        for (int j = 0; j < 8; ++j) {
          int M = j + hi * 8;
          dst[M * 32 + t0 + lo16] = c[j];
        }
      }
    }
  }
  __syncthreads();

  // ---- Phase C: S = (Qs*K^T)*wh + rel_k bias, masked, to LDS ----
  // K loaded UNSCALED; head weight applied to the 8 C elements instead.
  for (int cp = wv; cp < NPAIR; cp += 4) {
    {
      int npm = cp * 32 + 128 + lo16;   // prefetch next pair's K row
      if (npm <= 576)
        __builtin_prefetch(x + ((size_t)b * NTOK + npm) * CD + (NHEADS + h) * HD, 0, 3);
    }
#pragma unroll
    for (int half = 0; half < 2; ++half) {
      int m0    = cp * 32 + half * 16;
      int mcol  = m0 + lo16;
      int mload = mcol > 576 ? 576 : mcol;
      const float* kbase = x + ((size_t)b * NTOK + mload) * CD + (NHEADS + h) * HD;
      v8f c = {0.f, 0.f, 0.f, 0.f, 0.f, 0.f, 0.f, 0.f};
#pragma unroll
      for (int ks = 0; ks < 2; ++ks) {
        const float* kb = kbase + ks * 32 + hi * 16;   // contiguous K-run
        v16bf bk;
#pragma unroll
        for (int e = 0; e < 16; ++e) bk[e] = (__bf16)kb[e];
        c = wmma_bf16(ks == 0 ? aq0 : aq1, bk, c);
      }
#pragma unroll
      for (int j = 0; j < 8; ++j) {
        int M = j + hi * 8;
        int n = i0 + M;
        float val;
        if (mcol >= NTOK) {
          val = -1e30f;
        } else {
          int iv, ih2; rel_iv_ih(n, mcol, iv, ih2);
          val = c[j] * wh + qtv[M * 32 + iv] + qth[M * 32 + ih2];
        }
        s_lds[M * COLS + mcol] = val;
      }
    }
  }
  __syncthreads();

  // ---- Phase D: row softmax (unnormalized) + index histograms ----
  {
    int r = tid >> 3, sub = tid & 7;
    int mbeg = sub * (COLS / 8), mend = mbeg + (COLS / 8);  // contiguous chunk
    float pm = -3.0e38f;
    for (int m = mbeg; m < mend; ++m) pm = fmaxf(pm, s_lds[r * COLS + m]);
    red[r * 8 + sub] = pm;
    __syncthreads();
    float rmax = red[r * 8];
#pragma unroll
    for (int i = 1; i < 8; ++i) rmax = fmaxf(rmax, red[r * 8 + i]);
    __syncthreads();
    float ps = 0.0f;
    int n = i0 + r;
    for (int m = mbeg; m < mend; ++m) {
      float e = __expf(s_lds[r * COLS + m] - rmax);
      s_lds[r * COLS + m] = e;
      ps += e;
      int iv, ih2; rel_iv_ih(n, m, iv, ih2);
      atomicAdd(&av[r * 32 + iv],  e);   // ds_add_f32; masked cols add 0
      atomicAdd(&ah[r * 32 + ih2], e);
    }
    red[r * 8 + sub] = ps;
    __syncthreads();
    if (sub == 0) {
      float s = 0.0f;
#pragma unroll
      for (int i = 0; i < 8; ++i) s += red[r * 8 + i];
      rowsum[r] = s;
    }
  }
  __syncthreads();

  // ---- Phase E: O_partial = P * V (V unscaled; wh applied once at store) ----
  v8f oacc[4];
#pragma unroll
  for (int dt = 0; dt < 4; ++dt) {
    v8f z = {0.f, 0.f, 0.f, 0.f, 0.f, 0.f, 0.f, 0.f};
    oacc[dt] = z;
  }
  for (int cp = wv; cp < NPAIR; cp += 4) {
    int m0p = cp * 32;
    {
      int npm = m0p + 128 + lane;       // prefetch next pair's V row
      if (npm <= 576)
        __builtin_prefetch(x + ((size_t)b * NTOK + npm) * CD + (2 * NHEADS + h) * HD, 0, 3);
    }
    v16bf ap;                                     // A frag of P (16x32)
    {
      const float* p0 = &s_lds[lo16 * COLS + m0p + hi * 16];
#pragma unroll
      for (int e = 0; e < 8; ++e) {
        ap[e]     = (__bf16)p0[e];
        ap[e + 8] = (__bf16)p0[e + 8];
      }
    }
#pragma unroll
    for (int dt = 0; dt < 4; ++dt) {
      int d = dt * 16 + lo16;
      v16bf bv;                                   // B frag of V (32x16)
#pragma unroll
      for (int e = 0; e < 16; ++e) {
        int m = m0p + hi * 16 + e;
        if (m > 576) m = 576;                     // P is 0 there anyway
        bv[e] = (__bf16)x[((size_t)b * NTOK + m) * CD + (2 * NHEADS + h) * HD + d];
      }
      oacc[dt] = wmma_bf16(ap, bv, oacc[dt]);
    }
  }
#pragma unroll
  for (int dt = 0; dt < 4; ++dt) {
#pragma unroll
    for (int j = 0; j < 8; ++j) {
      int M = j + hi * 8;
      opart[wv * (16 * HD) + M * HD + dt * 16 + lo16] = oacc[dt][j] * wh;
    }
  }
  __syncthreads();

  // ---- Phase F: rel_v reconstruction: Av@Vv + Ah@Vh (wave wv -> d-tile wv) ----
  {
    int dt = wv;
    int d  = dt * 16 + lo16;
    v16bf aAv, aAh;
    {
      const float* pv = &av[lo16 * 32 + hi * 16];
      const float* ph = &ah[lo16 * 32 + hi * 16];
#pragma unroll
      for (int e = 0; e < 8; ++e) {
        aAv[e] = (__bf16)pv[e];  aAv[e + 8] = (__bf16)pv[e + 8];
        aAh[e] = (__bf16)ph[e];  aAh[e + 8] = (__bf16)ph[e + 8];
      }
    }
    v16bf bVv, bVh;
#pragma unroll
    for (int e = 0; e < 16; ++e) {
      int   t   = hi * 16 + e;
      int   tc  = (t < TT) ? t : 0;            // branch-free masked gather
      float msk = (t < TT) ? 1.0f : 0.0f;
      bVv[e] = (__bf16)(tvv[tc * HD + d] * msk);
      bVh[e] = (__bf16)(tvh[tc * HD + d] * msk);
    }
    v8f cr = {0.f, 0.f, 0.f, 0.f, 0.f, 0.f, 0.f, 0.f};
    cr = wmma_bf16(aAv, bVv, cr);
    cr = wmma_bf16(aAh, bVh, cr);
#pragma unroll
    for (int j = 0; j < 8; ++j) {
      int M = j + hi * 8;
      opart[M * HD + d] += cr[j];   // wave-exclusive column range, post-barrier
    }
  }
  __syncthreads();

  // ---- Phase G: reduce partials, normalize by rowsum, store ----
  {
    int r  = tid >> 3;
    int c0 = (tid & 7) * 8;
    int n  = i0 + r;
    if (n < NTOK) {
      float inv = 1.0f / rowsum[r];
      float* op = out + ((size_t)b * NTOK + n) * (NHEADS * HD) + h * HD + c0;
#pragma unroll
      for (int c = 0; c < 8; ++c) {
        float s = 0.0f;
#pragma unroll
        for (int wq = 0; wq < 4; ++wq) s += opart[wq * (16 * HD) + r * HD + c0 + c];
        op[c] = s * inv;
      }
    }
  }
}

extern "C" void kernel_launch(void* const* d_in, const int* in_sizes, int n_in,
                              void* d_out, int out_size, void* d_ws, size_t ws_size,
                              hipStream_t stream) {
  (void)in_sizes; (void)n_in; (void)out_size; (void)d_ws; (void)ws_size;
  const float* x   = (const float*)d_in[0];
  const float* wts = (const float*)d_in[1];
  const float* tkv = (const float*)d_in[2];
  const float* tkh = (const float*)d_in[3];
  const float* tvv = (const float*)d_in[4];
  const float* tvh = (const float*)d_in[5];
  float* out = (float*)d_out;
  dim3 grid(37, NHEADS, 16);   // 16-row q tiles x heads x batch
  HeadMaskAttnMixture_kernel<<<grid, dim3(128), 0, stream>>>(x, wts, tkv, tkh, tvv, tvh, out);
}
